// GraphAttentionLayer_67035849556311
// MI455X (gfx1250) — matrix operand order repaired and checked by reading, not compiled, not run on hardware
//
#include <hip/hip_runtime.h>
#include <hip/hip_bf16.h>

typedef __attribute__((ext_vector_type(16))) _Float16 v16h;
typedef __attribute__((ext_vector_type(8)))  float    v8f;
typedef __attribute__((ext_vector_type(4)))  int      v4i_t;

#define BATCH 4
#define NN    4096
#define FF    128
#define ALPHA 0.1f
#define LOG2E 1.4426950408889634f

// workspace offsets (in 32-bit words)
#define C1_OFF   0
#define C2_OFF   128
#define DD_OFF   256
#define M2_OFF   272
#define WH1_OFF  1024
#define WH2_OFF  (WH1_OFF + BATCH*NN)          // 17408
#define VP_OFF   (WH2_OFF + BATCH*NN)          // 33792  (packed-f16 vp, dwords)
#define H_OFF    (VP_OFF + (BATCH*NN*FF/2))    // 1082368 (f32 h)

#if defined(__gfx1250__) && __has_builtin(__builtin_amdgcn_global_load_async_to_lds_b128)
#define HAVE_ASYNC_LDS 1
#else
#define HAVE_ASYNC_LDS 0
#endif

#if defined(__gfx1250__) && __has_builtin(__builtin_amdgcn_s_wait_asynccnt)
#define WAIT_ASYNC(n) __builtin_amdgcn_s_wait_asynccnt(n)
#elif defined(__gfx1250__)
#define WAIT_ASYNC(n) asm volatile("s_wait_asynccnt %0" :: "i"(n) : "memory")
#else
#define WAIT_ASYNC(n) do {} while (0)
#endif

__device__ __forceinline__ float leaky_f(float x) { return x >= 0.f ? x : ALPHA * x; }

// ---------------------------------------------------------------------------
// Kernel 1: fold attention vector 'a' into the q/k projections.
// c1 = Wq_w^T a[:F], d1 = Wq_b . a[:F]   (same for k with a[F:])
// ---------------------------------------------------------------------------
__global__ void k_fold(const float* __restrict__ Wq_w, const float* __restrict__ Wq_b,
                       const float* __restrict__ Wk_w, const float* __restrict__ Wk_b,
                       const float* __restrict__ a, float* __restrict__ ws) {
  __shared__ float red[128];
  const int c = threadIdx.x;
  float s1 = 0.f, s2 = 0.f;
  for (int f = 0; f < FF; ++f) {
    s1 += Wq_w[f * FF + c] * a[f];
    s2 += Wk_w[f * FF + c] * a[FF + f];
  }
  ws[C1_OFF + c] = s1;
  ws[C2_OFF + c] = s2;

  red[c] = Wq_b[c] * a[c];
  __syncthreads();
  for (int st = 64; st > 0; st >>= 1) { if (c < st) red[c] += red[c + st]; __syncthreads(); }
  if (c == 0) ws[DD_OFF] = red[0];
  __syncthreads();
  red[c] = Wk_b[c] * a[FF + c];
  __syncthreads();
  for (int st = 64; st > 0; st >>= 1) { if (c < st) red[c] += red[c + st]; __syncthreads(); }
  if (c == 0) ws[DD_OFF + 1] = red[0];
}

// ---------------------------------------------------------------------------
// Kernel 2: Wh1[row] = q[row,:] . c1 + d1 ; Wh2[row] = k[row,:] . c2 + d2
// ---------------------------------------------------------------------------
__global__ void k_wh(const float* __restrict__ q, const float* __restrict__ k,
                     float* __restrict__ ws) {
  const int wave = threadIdx.x >> 5, lane = threadIdx.x & 31;
  const int row  = blockIdx.x * 8 + wave;
  const bool is2 = (blockIdx.y == 1);
  const float* X = is2 ? k : q;
  const float* c = ws + (is2 ? C2_OFF : C1_OFF);
  const float  d = ws[DD_OFF + (is2 ? 1 : 0)];
  const float4 x  = ((const float4*)(X + (size_t)row * FF))[lane];
  const float4 cc = ((const float4*)c)[lane];
  float p = x.x * cc.x + x.y * cc.y + x.z * cc.z + x.w * cc.w;
  #pragma unroll
  for (int m = 16; m >= 1; m >>= 1) p += __shfl_xor(p, m, 32);
  if (lane == 0) ws[(is2 ? WH2_OFF : WH1_OFF) + row] = p + d;
}

// ---------------------------------------------------------------------------
// Kernel 2b: per-batch max of Wh2 (softmax shift bound; leaky is monotone).
// ---------------------------------------------------------------------------
__global__ void k_m2(float* __restrict__ ws) {
  __shared__ float red[256];
  const int b = blockIdx.x;
  const float* w2 = ws + WH2_OFF + (size_t)b * NN;
  float m = -1e30f;
  for (int i = threadIdx.x; i < NN; i += 256) m = fmaxf(m, w2[i]);
  red[threadIdx.x] = m;
  __syncthreads();
  for (int st = 128; st > 0; st >>= 1) {
    if (threadIdx.x < st) red[threadIdx.x] = fmaxf(red[threadIdx.x], red[threadIdx.x + st]);
    __syncthreads();
  }
  if (threadIdx.x == 0) ws[M2_OFF + b] = red[0];
}

// ---------------------------------------------------------------------------
// Kernel 3: vp = v @ Wv_w^T + Wv_b  (WMMA f16), packed into the WMMA B-operand
// layout (lane = column, K node-pairs per VGPR) via an LDS shuffle.
// ---------------------------------------------------------------------------
__global__ void k_vp(const float* __restrict__ v, const float* __restrict__ Wv_w,
                     const float* __restrict__ Wv_b, float* __restrict__ ws) {
  __shared__ float lds[8][32][16];
  const int wave = threadIdx.x >> 5;
  const int lane = threadIdx.x & 31;
  const int half = lane >> 4;
  const int ln   = lane & 15;
  const int j0   = blockIdx.x * 32;   // global node base over B*N
  const int f0   = wave * 16;
  const int kb    = half ? 8 : 0;
  const int koffB = half ? 16 : 0;

  v8f acc0 = {0.f,0.f,0.f,0.f,0.f,0.f,0.f,0.f};
  v8f acc1 = {0.f,0.f,0.f,0.f,0.f,0.f,0.f,0.f};

  #pragma unroll
  for (int kt = 0; kt < 4; ++kt) {
    const int k0 = kt * 32;
    const float* wr = Wv_w + (size_t)(f0 + ln) * FF + k0 + kb;
    v16h bm;
    #pragma unroll
    for (int e = 0; e < 16; ++e) bm[e] = (_Float16)wr[e + (e & 8)];
    const float* x0 = v + (size_t)(j0 + ln) * FF + k0 + kb;
    const float* x1 = v + (size_t)(j0 + 16 + ln) * FF + k0 + kb;
    v16h a0, a1;
    #pragma unroll
    for (int e = 0; e < 8; ++e) { a0[e]   = (_Float16)x0[e];      a1[e]   = (_Float16)x1[e]; }
    #pragma unroll
    for (int e = 0; e < 8; ++e) { a0[8+e] = (_Float16)x0[16 + e]; a1[8+e] = (_Float16)x1[16 + e]; }
    acc0 = __builtin_amdgcn_wmma_f32_16x16x32_f16(false, a0, false, bm, (short)0, acc0, false, false);
    acc1 = __builtin_amdgcn_wmma_f32_16x16x32_f16(false, a1, false, bm, (short)0, acc1, false, false);
  }

  const float bias = Wv_b[f0 + ln];
  #pragma unroll
  for (int r = 0; r < 8; ++r) {
    lds[wave][     8 * half + r][ln] = acc0[r] + bias;
    lds[wave][16 + 8 * half + r][ln] = acc1[r] + bias;
  }
  __syncthreads();

  unsigned d[8];
  #pragma unroll
  for (int vv = 0; vv < 8; ++vv) {
    _Float16 lo = (_Float16)lds[wave][kb + 2 * vv + ((vv & 4) << 1)    ][ln];
    _Float16 hi = (_Float16)lds[wave][kb + 2 * vv + ((vv & 4) << 1) + 1][ln];
    d[vv] = (unsigned)__builtin_bit_cast(unsigned short, lo)
          | ((unsigned)__builtin_bit_cast(unsigned short, hi) << 16);
  }
  unsigned* pack = (unsigned*)ws + VP_OFF
                 + ((size_t)blockIdx.x * 8 + wave) * 256 + (size_t)lane * 8;
  uint4* p4 = (uint4*)pack;
  p4[0] = make_uint4(d[0], d[1], d[2], d[3]);
  p4[1] = make_uint4(d[4], d[5], d[6], d[7]);
}

// ---------------------------------------------------------------------------
// Kernel 4: fused masked-softmax attention:  h = softmax(mask(e)) @ vp
// Rank-1 logits; m_i = leaky(Wh1_i + max(Wh2)) is a valid shift => no online
// rescale. B tiles of vp are shared by all 4 waves of the block: staged once
// per K-step into LDS with async global->LDS DMA (ASYNCcnt), double-buffered
// so the DMA for step jt+1 overlaps the exp/WMMA work of step jt.
// ---------------------------------------------------------------------------
__global__ void k_att(const int* __restrict__ adj, float* __restrict__ ws) {
#if HAVE_ASYNC_LDS
  __shared__ unsigned lds_vp[2][2048];   // 2 x 8KB: 8 B-tiles (ft=0..7) per buffer
#endif
  const int wave = threadIdx.x >> 5;
  const int lane = threadIdx.x & 31;
  const int half = lane >> 4;
  const int ln   = lane & 15;
  const int gi0  = blockIdx.x * 64 + wave * 16;  // global row base over B*N
  const int b    = gi0 >> 12;                    // / NN
  const int kb   = half ? 8 : 0;

  const float wh1v = ws[WH1_OFF + gi0 + ln];
  const float M2   = ws[M2_OFF + b];
  const float mrow = leaky_f(wh1v + M2);
  const float* wh2 = ws + WH2_OFF + (size_t)b * NN;
  const int* adjrow = adj + ((size_t)gi0 + ln) * NN;
  const unsigned* vpbase = (const unsigned*)ws + VP_OFF + (size_t)b * 128 * 8 * 256;

#if HAVE_ASYNC_LDS
  // wave w stages ft-tiles (2w, 2w+1): 512 dwords = 4 async b128 per wave.
  auto stage = [&](int j, int buf) {
    const unsigned* src = vpbase + (size_t)j * 2048 + wave * 512 + lane * 4;
    unsigned* dst = &lds_vp[buf][wave * 512 + lane * 4];
    #pragma unroll
    for (int i = 0; i < 4; ++i)
      __builtin_amdgcn_global_load_async_to_lds_b128(
          (__attribute__((address_space(1))) v4i_t*)(src + i * 128),
          (__attribute__((address_space(3))) v4i_t*)(dst + i * 128), 0, 0);
  };
  stage(0, 0);
#endif

  v8f acc[8];
  #pragma unroll
  for (int f = 0; f < 8; ++f) acc[f] = v8f{0.f,0.f,0.f,0.f,0.f,0.f,0.f,0.f};
  float lsum = 0.f;

  for (int jt = 0; jt < 128; ++jt) {
    const int j0 = jt * 32;
    __builtin_prefetch(adjrow + j0 + 512, 0, 0);   // speculative, stream adj ahead
    const int4   A0 = *(const int4*)  (adjrow + j0 + kb);
    const int4   A1 = *(const int4*)  (adjrow + j0 + kb + 4);
    const int4   A2 = *(const int4*)  (adjrow + j0 + kb + 16);
    const int4   A3 = *(const int4*)  (adjrow + j0 + kb + 20);
    const float4 W0 = *(const float4*)(wh2 + j0 + kb);
    const float4 W1 = *(const float4*)(wh2 + j0 + kb + 4);
    const float4 W2 = *(const float4*)(wh2 + j0 + kb + 16);
    const float4 W3 = *(const float4*)(wh2 + j0 + kb + 20);

    v16h am;
    auto pe = [&](int ad, float w2v) -> _Float16 {
      float ee = leaky_f(wh1v + w2v);
      float p  = (ad > 0) ? exp2f((ee - mrow) * LOG2E) : 0.f;   // exponent <= 0
      lsum += p;
      return (_Float16)p;
    };
    am[0]  = pe(A0.x, W0.x); am[1]  = pe(A0.y, W0.y); am[2]  = pe(A0.z, W0.z); am[3]  = pe(A0.w, W0.w);
    am[4]  = pe(A1.x, W1.x); am[5]  = pe(A1.y, W1.y); am[6]  = pe(A1.z, W1.z); am[7]  = pe(A1.w, W1.w);
    am[8]  = pe(A2.x, W2.x); am[9]  = pe(A2.y, W2.y); am[10] = pe(A2.z, W2.z); am[11] = pe(A2.w, W2.w);
    am[12] = pe(A3.x, W3.x); am[13] = pe(A3.y, W3.y); am[14] = pe(A3.z, W3.z); am[15] = pe(A3.w, W3.w);

#if HAVE_ASYNC_LDS
    __syncthreads();                       // everyone done reading buf[(jt+1)&1] last step
    if (jt + 1 < 128) { stage(jt + 1, (jt + 1) & 1); WAIT_ASYNC(4); }
    else              { WAIT_ASYNC(0); }
    __syncthreads();                       // current buffer complete block-wide
    const unsigned* bufp = &lds_vp[jt & 1][0];
    #pragma unroll
    for (int ft = 0; ft < 8; ++ft) {
      const uint4* bq = (const uint4*)(bufp + ft * 256 + lane * 8);
      union { uint4 u[2]; v16h h; } cv;
      cv.u[0] = bq[0]; cv.u[1] = bq[1];
      acc[ft] = __builtin_amdgcn_wmma_f32_16x16x32_f16(false, am, false, cv.h,
                                                       (short)0, acc[ft], false, false);
    }
#else
    #pragma unroll
    for (int ft = 0; ft < 8; ++ft) {
      const uint4* bq = (const uint4*)(vpbase + (size_t)(jt * 8 + ft) * 256 + (size_t)lane * 8);
      union { uint4 u[2]; v16h h; } cv;
      cv.u[0] = bq[0]; cv.u[1] = bq[1];
      acc[ft] = __builtin_amdgcn_wmma_f32_16x16x32_f16(false, am, false, cv.h,
                                                       (short)0, acc[ft], false, false);
    }
#endif
  }

  // row sums: row ln split across lanes (ln, ln+16)
  lsum += __shfl_xor(lsum, 16, 32);
  const float inv = 1.0f / fmaxf(lsum, 1e-30f);
  float invr[8];
  #pragma unroll
  for (int r = 0; r < 8; ++r) invr[r] = __shfl(inv, 8 * half + r, 32);

  float* h = ws + H_OFF;
  #pragma unroll
  for (int ft = 0; ft < 8; ++ft) {
    #pragma unroll
    for (int r = 0; r < 8; ++r)
      h[(size_t)(gi0 + 8 * half + r) * FF + ft * 16 + ln] = acc[ft][r] * invr[r];
  }
}

// ---------------------------------------------------------------------------
// Kernel 5: out = leaky(h @ Wo_w^T)   (WMMA f16, f32 accumulate)
// ---------------------------------------------------------------------------
__global__ void k_out(const float* __restrict__ Wo_w, const float* __restrict__ ws,
                      float* __restrict__ out) {
  const float* h = ws + H_OFF;
  const int wave = threadIdx.x >> 5;
  const int lane = threadIdx.x & 31;
  const int half = lane >> 4;
  const int ln   = lane & 15;
  const int g0   = blockIdx.x * 32;
  const int f0   = wave * 16;
  const int kb    = half ? 8 : 0;
  const int koffB = half ? 16 : 0;

  v8f acc0 = {0.f,0.f,0.f,0.f,0.f,0.f,0.f,0.f};
  v8f acc1 = {0.f,0.f,0.f,0.f,0.f,0.f,0.f,0.f};

  #pragma unroll
  for (int kt = 0; kt < 4; ++kt) {
    const int k0 = kt * 32;
    const float* wr = Wo_w + (size_t)(f0 + ln) * FF + k0 + kb;
    v16h bm;
    #pragma unroll
    for (int e = 0; e < 16; ++e) bm[e] = (_Float16)wr[e + (e & 8)];
    const float* x0 = h + (size_t)(g0 + ln) * FF + k0 + kb;
    const float* x1 = h + (size_t)(g0 + 16 + ln) * FF + k0 + kb;
    v16h a0, a1;
    #pragma unroll
    for (int e = 0; e < 8; ++e) { a0[e]   = (_Float16)x0[e];      a1[e]   = (_Float16)x1[e]; }
    #pragma unroll
    for (int e = 0; e < 8; ++e) { a0[8+e] = (_Float16)x0[16 + e]; a1[8+e] = (_Float16)x1[16 + e]; }
    acc0 = __builtin_amdgcn_wmma_f32_16x16x32_f16(false, a0, false, bm, (short)0, acc0, false, false);
    acc1 = __builtin_amdgcn_wmma_f32_16x16x32_f16(false, a1, false, bm, (short)0, acc1, false, false);
  }

  #pragma unroll
  for (int r = 0; r < 8; ++r) {
    out[(size_t)(g0 +      8 * half + r) * FF + f0 + ln] = leaky_f(acc0[r]);
    out[(size_t)(g0 + 16 + 8 * half + r) * FF + f0 + ln] = leaky_f(acc1[r]);
  }
}

// ---------------------------------------------------------------------------
extern "C" void kernel_launch(void* const* d_in, const int* in_sizes, int n_in,
                              void* d_out, int out_size, void* d_ws, size_t ws_size,
                              hipStream_t stream) {
  const float* q    = (const float*)d_in[0];
  const float* k    = (const float*)d_in[1];
  const float* v    = (const float*)d_in[2];
  const int*   adj  = (const int*)  d_in[3];
  const float* Wq_w = (const float*)d_in[4];
  const float* Wq_b = (const float*)d_in[5];
  const float* Wk_w = (const float*)d_in[6];
  const float* Wk_b = (const float*)d_in[7];
  const float* Wv_w = (const float*)d_in[8];
  const float* Wv_b = (const float*)d_in[9];
  const float* a    = (const float*)d_in[10];
  const float* Wo_w = (const float*)d_in[11];
  float* out = (float*)d_out;
  float* ws  = (float*)d_ws;

  k_fold<<<1, 128, 0, stream>>>(Wq_w, Wq_b, Wk_w, Wk_b, a, ws);
  k_wh<<<dim3((BATCH * NN) / 8, 2), 256, 0, stream>>>(q, k, ws);
  k_m2<<<BATCH, 256, 0, stream>>>(ws);
  k_vp<<<(BATCH * NN) / 32, 256, 0, stream>>>(v, Wv_w, Wv_b, ws);
  k_att<<<(BATCH * NN) / 64, 128, 0, stream>>>(adj, ws);
  k_out<<<(BATCH * NN) / 32, 256, 0, stream>>>(Wo_w, ws, out);
}
